// SelfOuterAttention_14181982012024
// MI455X (gfx1250) — compile-verified, matrix operand order checked by reading, not run on hardware
//
#include <hip/hip_runtime.h>
#include <hip/hip_bf16.h>

typedef __attribute__((ext_vector_type(16))) _Float16 v16h;
typedef __attribute__((ext_vector_type(8)))  float    v8f;

#define Bsz 4
#define Nn  512
#define Dd  128
#define BND (Bsz * Nn * Dd)

// Load 16 f16 elements for one lane of a 16x16x32 WMMA operand.
// base points at row_start + k0 + (hi ? 8 : 0); elements 0..7 = base[0..7],
// elements 8..15 = base[16..23] (per the 16-bit A/B VGPR layout).
__device__ __forceinline__ v16h load_op16(const float* __restrict__ base) {
    const float4* p0 = (const float4*)(base);
    const float4* p1 = (const float4*)(base + 16);
    float4 a0 = p0[0], a1 = p0[1];
    float4 a2 = p1[0], a3 = p1[1];
    v16h r;
    r[0]  = (_Float16)a0.x; r[1]  = (_Float16)a0.y;
    r[2]  = (_Float16)a0.z; r[3]  = (_Float16)a0.w;
    r[4]  = (_Float16)a1.x; r[5]  = (_Float16)a1.y;
    r[6]  = (_Float16)a1.z; r[7]  = (_Float16)a1.w;
    r[8]  = (_Float16)a2.x; r[9]  = (_Float16)a2.y;
    r[10] = (_Float16)a2.z; r[11] = (_Float16)a2.w;
    r[12] = (_Float16)a3.x; r[13] = (_Float16)a3.y;
    r[14] = (_Float16)a3.z; r[15] = (_Float16)a3.w;
    return r;
}

// Y[m, e] = sum_d X[m, d] * W[e, d] + bias[e]    (torch Linear semantics)
// X: [M, 128] f32, W: [128, 128] f32 (row-major [out, in]), Y: [M, 128] f32.
// One wave32 computes one 16x16 tile of Y via v_wmma_f32_16x16x32_f16.
__global__ void __launch_bounds__(128)
gemm_wmma_f16(const float* __restrict__ X, const float* __restrict__ W,
              const float* __restrict__ bias, float* __restrict__ Y, int Mtiles) {
    const int lane = threadIdx.x & 31;
    const int wib  = threadIdx.x >> 5;
    const int wave = blockIdx.x * 4 + wib;
    const int tm = wave >> 3;          // 8 column tiles (128/16)
    const int tn = wave & 7;
    if (tm >= Mtiles) return;          // uniform across the wave

    const int r   = lane & 15;
    const int hi  = lane >> 4;         // 0 or 1
    const int row = tm * 16 + r;       // A-matrix row for this lane
    const int col = tn * 16 + r;       // B-matrix column for this lane

    v8f acc = {0.f, 0.f, 0.f, 0.f, 0.f, 0.f, 0.f, 0.f};
    #pragma unroll
    for (int k0 = 0; k0 < 128; k0 += 32) {
        v16h a = load_op16(X + row * Dd + k0 + hi * 8);
        v16h b = load_op16(W + col * Dd + k0 + hi * 8);
        acc = __builtin_amdgcn_wmma_f32_16x16x32_f16(
            /*neg_a=*/false, a, /*neg_b=*/false, b,
            /*c_mod=*/(short)0, acc, /*reuse_a=*/false, /*reuse_b=*/false);
    }

    // C/D layout: VGPR i -> (M = i + hi*8, N = lane&15)
    const float bc = bias[col];
    const int m0 = tm * 16 + hi * 8;
    #pragma unroll
    for (int i = 0; i < 8; ++i)
        Y[(m0 + i) * Dd + col] = acc[i] + bc;
}

// Per-channel outer-product attention, fused (no score materialization).
// out[b,n,d] = sum_m softmax_m(qp[b,n,d]*kp[b,m,d]) * vp[b,m,d]
// Stabilization: rowmax = q>0 ? q*max_m(K) : q*min_m(K).
// mask[b,n]==0 => all scores equal => uniform softmax == setting q=0.
// Layout: lane <-> channel d (coalesced), 4 queries per thread.
__global__ void __launch_bounds__(256)
outer_attn(const float* __restrict__ qp, const float* __restrict__ kp,
           const float* __restrict__ vp, const int* __restrict__ mask,
           float* __restrict__ out) {
    const int d   = threadIdx.x & 127;
    const int sub = threadIdx.x >> 7;              // 0 or 1
    const int b   = blockIdx.x >> 6;               // 64 query-tiles of 8 per batch
    const int n0  = (blockIdx.x & 63) * 8 + sub * 4;

    const float* kb = kp + (size_t)b * Nn * Dd + d;
    const float* vb = vp + (size_t)b * Nn * Dd + d;

    // Pass 1: per-channel K min / max (coalesced, L2-resident).
    float kmx = -3.4e38f, kmn = 3.4e38f;
    for (int m = 0; m < Nn; ++m) {
        float kk = kb[m * Dd];
        kmx = fmaxf(kmx, kk);
        kmn = fminf(kmn, kk);
    }

    float q[4], rmax[4], num[4], den[4];
    #pragma unroll
    for (int i = 0; i < 4; ++i) {
        const int n = n0 + i;
        float qq = qp[((size_t)b * Nn + n) * Dd + d];
        if (mask[b * Nn + n] == 0) qq = 0.f;       // uniform-softmax equivalence
        q[i]    = qq;
        rmax[i] = (qq > 0.f) ? qq * kmx : qq * kmn;
        num[i]  = 0.f;
        den[i]  = 0.f;
    }

    // Pass 2: streaming softmax accumulation over keys.
    for (int m = 0; m < Nn; ++m) {
        if (m + 32 < Nn)
            __builtin_prefetch(kb + (m + 32) * Dd, 0, 1);  // global_prefetch_b8
        const float kk = kb[m * Dd];
        const float vv = vb[m * Dd];
        #pragma unroll
        for (int i = 0; i < 4; ++i) {
            const float e = __expf(fmaf(q[i], kk, -rmax[i]));
            num[i] = fmaf(e, vv, num[i]);
            den[i] += e;
        }
    }

    #pragma unroll
    for (int i = 0; i < 4; ++i)
        out[((size_t)b * Nn + n0 + i) * Dd + d] = num[i] / den[i];
}

extern "C" void kernel_launch(void* const* d_in, const int* in_sizes, int n_in,
                              void* d_out, int out_size, void* d_ws, size_t ws_size,
                              hipStream_t stream) {
    const float* q    = (const float*)d_in[0];
    const float* k    = (const float*)d_in[1];
    const float* v    = (const float*)d_in[2];
    const int*   mask = (const int*)  d_in[3];
    const float* Wq   = (const float*)d_in[4];
    const float* bq   = (const float*)d_in[5];
    const float* Wk   = (const float*)d_in[6];
    const float* bk   = (const float*)d_in[7];
    const float* Wv   = (const float*)d_in[8];
    const float* bv   = (const float*)d_in[9];
    const float* Wo   = (const float*)d_in[10];
    const float* bo   = (const float*)d_in[11];
    float* out = (float*)d_out;

    float* ws   = (float*)d_ws;
    float* qp   = ws;
    float* kpp  = ws + BND;
    float* vpp  = ws + 2 * BND;
    float* attn = ws + 3 * BND;

    const int Mtiles = (Bsz * Nn) / 16;            // 128 row tiles
    const int gemm_blocks = (Mtiles * 8 + 3) / 4;  // 4 waves (tiles) per block

    gemm_wmma_f16<<<gemm_blocks, 128, 0, stream>>>(q, Wq, bq, qp,  Mtiles);
    gemm_wmma_f16<<<gemm_blocks, 128, 0, stream>>>(k, Wk, bk, kpp, Mtiles);
    gemm_wmma_f16<<<gemm_blocks, 128, 0, stream>>>(v, Wv, bv, vpp, Mtiles);

    outer_attn<<<Bsz * 64, 256, 0, stream>>>(qp, kpp, vpp, mask, attn);

    gemm_wmma_f16<<<gemm_blocks, 128, 0, stream>>>(attn, Wo, bo, out, Mtiles);
}
